// HyenaOperator_77300821393772
// MI455X (gfx1250) — matrix operand order repaired and checked by reading, not compiled, not run on hardware
//
#include <hip/hip_runtime.h>

typedef __bf16 bf16;
typedef __attribute__((ext_vector_type(16))) __bf16 v16bf;
typedef __attribute__((ext_vector_type(8)))  __bf16 v8bf;
typedef __attribute__((ext_vector_type(8)))  float  v8f;

constexpr int Bn = 2, L = 4096, D = 2048, D3 = 6144, ORD = 64;
constexpr int M_TOT = Bn * L;           // 8192
constexpr int NFFT = 8192;              // 2*L

// ------------------------------------------------------------ CDNA5 async-LDS
// GLOBAL_LOAD_ASYNC_TO_LDS_B128: DMA 16B global -> LDS per lane, tracked by ASYNCcnt.
__device__ __forceinline__ void async_load_lds_b128(unsigned lds_off, const void* gptr) {
    asm volatile("global_load_async_to_lds_b128 %0, %1, off"
                 :: "v"(lds_off), "v"(gptr) : "memory");
}
__device__ __forceinline__ void wait_asynccnt0() {
    asm volatile("s_wait_asynccnt 0x0" ::: "memory");
}
__device__ __forceinline__ unsigned lds_off_of(const void* p) {
    // generic LDS addresses carry the LDS byte offset in the low 32 bits
    return (unsigned)(uintptr_t)p;
}

// ---------------------------------------------------------------- conversions
__global__ __launch_bounds__(256)
void cvt_bf16_kernel(const float* __restrict__ s, bf16* __restrict__ d, int n) {
    int i = blockIdx.x * 256 + threadIdx.x;
    if (i < n) d[i] = (bf16)s[i];
}

// ---------------------------------------------------------------- WMMA GEMM
// C = A(MxK,bf16) * W(NxK,bf16)^T + bias(N)
// tmode 0: C row-major (M x N) f32
// tmode 1: C as (B, N, L) f32 with m = b*L + l  (transposed store for conv stage)
__global__ __launch_bounds__(256)
void gemm_bf16_wmma_kernel(const bf16* __restrict__ A, const bf16* __restrict__ W,
                           const float* __restrict__ bias, float* __restrict__ C,
                           int M, int N, int K, int tmode) {
    const int lane  = threadIdx.x & 31;
    const int wave  = threadIdx.x >> 5;
    const int wm    = wave & 1;               // 2 waves along M
    const int wn    = wave >> 1;              // 4 waves along N
    const int m_base = blockIdx.y * 64  + wm * 32;
    const int n_base = blockIdx.x * 128 + wn * 32;

    const int lhalf = (lane < 16) ? 0 : 1;
    const int lrow  = lane & 15;

    v8f acc00 = {}; v8f acc01 = {}; v8f acc10 = {}; v8f acc11 = {};

    const bf16* arow0 = A + (size_t)(m_base + lrow)      * K;
    const bf16* arow1 = A + (size_t)(m_base + 16 + lrow) * K;
    const bf16* brow0 = W + (size_t)(n_base + lrow)      * K;
    const bf16* brow1 = W + (size_t)(n_base + 16 + lrow) * K;

    for (int kb = 0; kb < K; kb += 32) {
        // A layout: per-lane 16 contiguous K (lanes0-15 K0-15, lanes16-31 K16-31)
        v8bf a0lo = *(const v8bf*)(arow0 + kb     + lhalf * 16);
        v8bf a0hi = *(const v8bf*)(arow0 + kb + 8 + lhalf * 16);
        v8bf a1lo = *(const v8bf*)(arow1 + kb     + lhalf * 16);
        v8bf a1hi = *(const v8bf*)(arow1 + kb + 8 + lhalf * 16);
        v16bf a0, a1;
        #pragma unroll
        for (int i = 0; i < 8; ++i) {
            a0[i] = a0lo[i]; a0[8 + i] = a0hi[i];
            a1[i] = a1lo[i]; a1[8 + i] = a1hi[i];
        }
        // B fragment: per-lane 16 contiguous K (lanes0-15 K0-15, lanes16-31 K16-31)
        v16bf b0 = *(const v16bf*)(brow0 + kb + lhalf * 16);
        v16bf b1 = *(const v16bf*)(brow1 + kb + lhalf * 16);

        acc00 = __builtin_amdgcn_wmma_f32_16x16x32_bf16(false, a0, false, b0, (short)0, acc00, false, false);
        acc01 = __builtin_amdgcn_wmma_f32_16x16x32_bf16(false, a0, false, b1, (short)0, acc01, false, false);
        acc10 = __builtin_amdgcn_wmma_f32_16x16x32_bf16(false, a1, false, b0, (short)0, acc10, false, false);
        acc11 = __builtin_amdgcn_wmma_f32_16x16x32_bf16(false, a1, false, b1, (short)0, acc11, false, false);
    }

    // D layout: vgpr r -> M = r (lanes 0-15) / 8+r (lanes 16-31), N = lane&15
    const int colN    = lrow;
    const int rowBase = lhalf * 8;
    #pragma unroll
    for (int r = 0; r < 8; ++r) {
        int mm0 = m_base + rowBase + r;
        int mm1 = m_base + 16 + rowBase + r;
        int nn0 = n_base + colN;
        int nn1 = n_base + 16 + colN;
        float v00 = acc00[r] + bias[nn0];
        float v01 = acc01[r] + bias[nn1];
        float v10 = acc10[r] + bias[nn0];
        float v11 = acc11[r] + bias[nn1];
        if (tmode == 0) {
            C[(size_t)mm0 * N + nn0] = v00;
            C[(size_t)mm0 * N + nn1] = v01;
            C[(size_t)mm1 * N + nn0] = v10;
            C[(size_t)mm1 * N + nn1] = v11;
        } else {
            int b0i = mm0 >> 12, l0i = mm0 & (L - 1);
            int b1i = mm1 >> 12, l1i = mm1 & (L - 1);
            size_t base0 = (size_t)b0i * N * L + l0i;
            size_t base1 = (size_t)b1i * N * L + l1i;
            C[base0 + (size_t)nn0 * L] = v00;
            C[base0 + (size_t)nn1 * L] = v01;
            C[base1 + (size_t)nn0 * L] = v10;
            C[base1 + (size_t)nn1 * L] = v11;
        }
    }
}

// ---------------------------------------------------------------- short conv
__global__ __launch_bounds__(256)
void shortconv_kernel(const float* __restrict__ ut, const float* __restrict__ sw,
                      const float* __restrict__ sb, float* __restrict__ x0,
                      float* __restrict__ wg) {
    int idx = blockIdx.x * 256 + threadIdx.x;   // over Bn*D*L
    int l  = idx & (L - 1);
    int bd = idx >> 12;
    int d  = bd & (D - 1);
    int b  = bd >> 11;
    float vals[3];
    #pragma unroll
    for (int g = 0; g < 3; ++g) {
        int c = g * D + d;
        const float* u = ut + ((size_t)b * D3 + c) * L;
        float um2 = (l >= 2) ? u[l - 2] : 0.f;
        float um1 = (l >= 1) ? u[l - 1] : 0.f;
        float u0  = u[l];
        vals[g] = um2 * sw[c * 3 + 0] + um1 * sw[c * 3 + 1] + u0 * sw[c * 3 + 2] + sb[c];
    }
    size_t o = ((size_t)b * D + d) * L + l;
    x0[o] = vals[0];
    wg[o] = vals[2] * vals[1];
}

// ---------------------------------------------------------------- filter k(D,L)
__global__ __launch_bounds__(256)
void filter_kernel(const float* __restrict__ w0, const float* __restrict__ b0,
                   const float* __restrict__ freq,
                   const float* __restrict__ w1, const float* __restrict__ b1,
                   const float* __restrict__ w2, const float* __restrict__ b2,
                   const float* __restrict__ w3, float* __restrict__ kout) {
    __shared__ float h[2][ORD];
    int l   = blockIdx.x;
    int tid = threadIdx.x;
    float t    = (float)l / (float)(L - 1);
    float wang = 6.2831853071795864f * (float)l / (float)L;
    float ang  = 1e-4f * wang;
    float z0 = t, z1 = cosf(ang), z2 = -sinf(ang);
    if (tid < ORD) {
        float a = z0 * w0[tid * 3 + 0] + z1 * w0[tid * 3 + 1] + z2 * w0[tid * 3 + 2] + b0[tid];
        h[0][tid] = sinf(freq[tid] * a);
    }
    __syncthreads();
    if (tid < ORD) {
        float a = b1[tid];
        for (int i = 0; i < ORD; ++i) a += h[0][i] * w1[tid * ORD + i];
        h[1][tid] = sinf(freq[tid] * a);
    }
    __syncthreads();
    if (tid < ORD) {
        float a = b2[tid];
        for (int i = 0; i < ORD; ++i) a += h[1][i] * w2[tid * ORD + i];
        h[0][tid] = sinf(freq[tid] * a);
    }
    __syncthreads();
    const float mind = -4.605170185988091f / 1.5f;  // log(0.01)/1.5
    const float maxd = -4.605170185988091f / 0.3f;  // log(0.01)/0.3
    for (int d = tid; d < D; d += 256) {
        float a = 0.f;
        for (int i = 0; i < ORD; ++i) a += h[0][i] * w3[d * ORD + i];
        float delta = mind + (maxd - mind) * (float)d / (float)(D - 1);
        float decay = __expf(-t * fabsf(delta));
        kout[(size_t)d * L + l] = a * decay;
    }
}

// ------------------------------------------- 8192-pt FFT in LDS, table twiddles
// tw[j] = W_8192^j = exp(-2*pi*i*j/8192), j in [0, 4096)
__device__ __forceinline__ void tw_init(float2* tw) {
    for (int j = threadIdx.x; j < NFFT / 2; j += 256) {
        float s, c;
        __sincosf(-6.2831853071795864f * (float)j / (float)NFFT, &s, &c);
        tw[j] = make_float2(c, s);
    }
}

__device__ __forceinline__ void fft8192_shared(float* re, float* im, const float2* tw) {
    const int tid = threadIdx.x;
    // bit-reverse permutation (13 bits)
    for (int i = tid; i < NFFT; i += 256) {
        int j = (int)(__brev((unsigned)i) >> 19);
        if (j > i) {
            float tr = re[i]; re[i] = re[j]; re[j] = tr;
            float ti = im[i]; im[i] = im[j]; im[j] = ti;
        }
    }
    __syncthreads();
    for (int len = 2; len <= NFFT; len <<= 1) {
        int half   = len >> 1;
        int stride = NFFT / len;
        for (int bi = tid; bi < NFFT / 2; bi += 256) {
            int grp = bi / half;
            int pos = bi - grp * half;
            int i0 = grp * len + pos;
            int i1 = i0 + half;
            float2 w = tw[pos * stride];
            float xr = re[i1], xi = im[i1];
            float tr = xr * w.x - xi * w.y;
            float ti = xr * w.y + xi * w.x;
            float ur = re[i0], ui = im[i0];
            re[i0] = ur + tr; im[i0] = ui + ti;
            re[i1] = ur - tr; im[i1] = ui - ti;
        }
        __syncthreads();
    }
}

// K_f = FFT(zero-pad(k)) / NFFT, full complex spectrum per channel
// dynamic LDS: re[8192] | im[8192] | tw[4096 float2]  = 96 KB
__global__ __launch_bounds__(256)
void fftk_kernel(const float* __restrict__ kin, float2* __restrict__ kf) {
    extern __shared__ float smem[];
    float*  re = smem;
    float*  im = smem + NFFT;
    float2* tw = (float2*)(smem + 2 * NFFT);

    int d = blockIdx.x;
    const float* kr = kin + (size_t)d * L;
    tw_init(tw);
    for (int i = threadIdx.x; i < NFFT; i += 256) {
        re[i] = (i < L) ? kr[i] : 0.f;
        im[i] = 0.f;
    }
    __syncthreads();
    fft8192_shared(re, im, tw);
    const float inv_n = 1.f / (float)NFFT;
    float2* o = kf + (size_t)d * NFFT;
    for (int i = threadIdx.x; i < NFFT; i += 256) {
        o[i] = make_float2(re[i] * inv_n, im[i] * inv_n);
    }
}

// y = Re(conj(FFT(conj(FFT(pad(wg)) * K_f))))[0:L] + wg*bias; gated = y * x0 -> bf16 (B*L, D)
// dynamic LDS: re[8192] | im[8192] | tw[4096 f2] | kfs[8192 f2] = 160 KB (2 WGs / 320KB WGP)
// K_f row (64 KB) is DMA'd LDS-ward with GLOBAL_LOAD_ASYNC_TO_LDS_B128 while the
// forward FFT runs; s_wait_asynccnt 0 + barrier before the pointwise multiply.
__global__ __launch_bounds__(256)
void fftconv_gate_kernel(const float* __restrict__ wg, const float* __restrict__ x0,
                         const float2* __restrict__ kf, const float* __restrict__ fbias,
                         bf16* __restrict__ gated) {
    extern __shared__ float smem[];
    float*  re  = smem;                          // 8192 f
    float*  im  = smem + NFFT;                   // 8192 f
    float2* tw  = (float2*)(smem + 2 * NFFT);    // 4096 f2
    float2* kfs = (float2*)(smem + 4 * NFFT);    // 8192 f2

    int d = blockIdx.x;
    int b = blockIdx.y;

    // kick off async DMA of this channel's spectrum into LDS (overlaps first FFT)
    const float2* kc = kf + (size_t)d * NFFT;
    for (int i = threadIdx.x * 2; i < NFFT; i += 512) {
        async_load_lds_b128(lds_off_of(&kfs[i]), &kc[i]);
    }

    const float* wr = wg + ((size_t)b * D + d) * L;
    tw_init(tw);
    for (int i = threadIdx.x; i < NFFT; i += 256) {
        re[i] = (i < L) ? wr[i] : 0.f;
        im[i] = 0.f;
    }
    __syncthreads();
    fft8192_shared(re, im, tw);

    wait_asynccnt0();        // this wave's DMA done
    __syncthreads();         // everyone's DMA done -> kfs valid
    for (int i = threadIdx.x; i < NFFT; i += 256) {
        float ur = re[i], ui = im[i];
        float2 kv = kfs[i];
        float pr = ur * kv.x - ui * kv.y;
        float pi = ur * kv.y + ui * kv.x;
        re[i] = pr; im[i] = -pi;           // conjugate -> inverse via forward FFT
    }
    __syncthreads();
    fft8192_shared(re, im, tw);

    const float* x0r = x0 + ((size_t)b * D + d) * L;
    float fb = fbias[d];
    for (int i = threadIdx.x; i < L; i += 256) {
        float yv = re[i] + wr[i] * fb;     // Re(conj(.)) = Re(.)
        float gv = yv * x0r[i];
        gated[((size_t)b * L + i) * D + d] = (bf16)gv;
    }
}

// ---------------------------------------------------------------- driver
extern "C" void kernel_launch(void* const* d_in, const int* in_sizes, int n_in,
                              void* d_out, int out_size, void* d_ws, size_t ws_size,
                              hipStream_t stream) {
    const float* x     = (const float*)d_in[0];
    const float* in_w  = (const float*)d_in[1];
    const float* in_b  = (const float*)d_in[2];
    const float* sw    = (const float*)d_in[3];
    const float* sb    = (const float*)d_in[4];
    const float* w0    = (const float*)d_in[5];
    const float* b0    = (const float*)d_in[6];
    const float* freq  = (const float*)d_in[7];
    const float* w1    = (const float*)d_in[8];
    const float* b1    = (const float*)d_in[9];
    const float* w2    = (const float*)d_in[10];
    const float* b2    = (const float*)d_in[11];
    const float* w3    = (const float*)d_in[12];
    const float* fbias = (const float*)d_in[13];
    const float* out_w = (const float*)d_in[14];
    const float* out_b = (const float*)d_in[15];
    float* out = (float*)d_out;

    char* ws = (char*)d_ws;
    bf16*   xb    = (bf16*)  (ws + (  0ull << 20));   // 32 MB: x bf16
    bf16*   wib   = (bf16*)  (ws + ( 32ull << 20));   // 24 MB: in_proj_w bf16
    bf16*   wob   = (bf16*)  (ws + ( 56ull << 20));   //  8 MB: out_proj_w bf16
    float*  ut    = (float*) (ws + ( 64ull << 20));   // 192 MB: u (B,3D,L)
    float*  x0b   = (float*) (ws + (256ull << 20));   // 64 MB: x0 (B,D,L)
    float*  wgb   = (float*) (ws + (320ull << 20));   // 64 MB: v*x1 (B,D,L)
    float*  kbuf  = (float*) (ws + (384ull << 20));   // 32 MB: k (D,L)
    float2* kfb   = (float2*)(ws + (416ull << 20));   // 128 MB: K_f (D, 8192) complex
    bf16*   gated = (bf16*)  (ws + (544ull << 20));   // 32 MB: gated (B*L, D) bf16

    // 1. fp32 -> bf16 for WMMA operands
    cvt_bf16_kernel<<<(M_TOT * D) / 256, 256, 0, stream>>>(x, xb, M_TOT * D);
    cvt_bf16_kernel<<<(D3 * D) / 256, 256, 0, stream>>>(in_w, wib, D3 * D);
    cvt_bf16_kernel<<<(D * D) / 256, 256, 0, stream>>>(out_w, wob, D * D);

    // 2. in_proj GEMM -> u transposed to (B, 3D, L)
    gemm_bf16_wmma_kernel<<<dim3(D3 / 128, M_TOT / 64), 256, 0, stream>>>(
        xb, wib, in_b, ut, M_TOT, D3, D, /*tmode=*/1);

    // 3. depthwise short conv + split + v*x1 gate
    shortconv_kernel<<<(Bn * D * L) / 256, 256, 0, stream>>>(ut, sw, sb, x0b, wgb);

    // 4. implicit filter k(D,L)
    filter_kernel<<<L, 256, 0, stream>>>(w0, b0, freq, w1, b1, w2, b2, w3, kbuf);

    // 5. K_f spectra (one WG per channel, whole FFT in LDS), 96 KB dynamic LDS
    fftk_kernel<<<D, 256, 96 * 1024, stream>>>(kbuf, kfb);

    // 6. FFT conv + bias + x0 gate, emit bf16 (B*L, D), 160 KB dynamic LDS
    fftconv_gate_kernel<<<dim3(D, Bn), 256, 160 * 1024, stream>>>(wgb, x0b, kfb, fbias, gated);

    // 7. out_proj GEMM -> final output (B, L, D) f32
    gemm_bf16_wmma_kernel<<<dim3(D / 128, M_TOT / 64), 256, 0, stream>>>(
        gated, wob, out_b, out, M_TOT, D, D, /*tmode=*/0);
}
